// SceneMultimodalCrossAttnTransformerLayer_21354577396027
// MI455X (gfx1250) — matrix operand order repaired and checked by reading, not compiled, not run on hardware
//
#include <hip/hip_runtime.h>
#include <cstdint>

// ---------------- problem constants ----------------
#define BB   4
#define HH   256
#define MM   4096          // A*T*NMODE queries per batch
#define NS   512           // scene tokens
#define NHEAD 8
#define DH   32            // head dim
#define FFD  128
#define LN_EPS 1e-6f

typedef __attribute__((ext_vector_type(16))) __bf16 v16bf;
typedef __attribute__((ext_vector_type(8)))  float  v8f;

// ----- WMMA fragment index helpers (16x16x32 bf16; A/B use contiguous k = 16*(lane>>4) + j) -----
__device__ __forceinline__ int a_k(int lane, int j) {        // A: lane holds row lane&15
    return ((lane >> 4) << 4) + j;
}
__device__ __forceinline__ int b_k(int lane, int j) {        // B: lane holds col lane&15
    return ((lane >> 4) << 4) + j;
}
__device__ __forceinline__ int c_row(int lane, int i) {      // C/D: element i -> row
    return i + ((lane >> 4) << 3);
}

__device__ __forceinline__ v8f wmma_bf16(v16bf a, v16bf b, v8f c) {
    return __builtin_amdgcn_wmma_f32_16x16x32_bf16(false, a, false, b, (short)0, c,
                                                   false, false);
}

// butterfly reductions across the 16 lanes of a half-wave (wave32)
__device__ __forceinline__ float redmax16(float v) {
#pragma unroll
    for (int m = 1; m < 16; m <<= 1) v = fmaxf(v, __shfl_xor(v, m, 32));
    return v;
}
__device__ __forceinline__ float redsum16(float v) {
#pragma unroll
    for (int m = 1; m < 16; m <<= 1) v += __shfl_xor(v, m, 32);
    return v;
}

// ----- CDNA5 async global->LDS copy (ASYNCcnt path), per ISA 08_async_tensor -----
__device__ __forceinline__ unsigned lds_off_u32(const void* p) {
    return (unsigned)(uintptr_t)p;           // low 32 bits of generic LDS ptr = LDS offset
}
__device__ __forceinline__ void async_ld_b128(unsigned lds_addr, uint64_t gaddr) {
    asm volatile("global_load_async_to_lds_b128 %0, %1, off"
                 :: "v"(lds_addr), "v"(gaddr) : "memory");
}
__device__ __forceinline__ void wait_async0() {
    asm volatile("s_wait_asynccnt 0x0" ::: "memory");
}

// cooperatively async-load 32 rows x 256 cols of bf16 (16 KB) into LDS
__device__ __forceinline__ void load_chunk_async(__bf16 (*dst)[HH],
                                                 const __bf16* __restrict__ src,
                                                 int tid) {
#pragma unroll
    for (int it = 0; it < 4; ++it) {
        const int idx = tid + it * 256;      // 1024 x 16-byte segments
        const int row = idx >> 5;
        const int seg = idx & 31;
        async_ld_b128(lds_off_u32(&dst[row][seg * 8]),
                      (uint64_t)(uintptr_t)(src + (size_t)row * HH + seg * 8));
    }
}

// =====================================================================
// Kernel 0: f32 -> bf16 weight pre-conversion (one-shot, removes per-block cvt)
// =====================================================================
__global__ __launch_bounds__(256) void cvt_bf16_kernel(const float* __restrict__ src,
                                                       __bf16* __restrict__ dst, int n)
{
    const int i = blockIdx.x * 256 + threadIdx.x;
    if (i < n) dst[i] = (__bf16)src[i];
}

// =====================================================================
// Kernel 1: projection  out_bf16 = (in @ W^T + bias) * scale   (W already bf16)
// block = 8 waves; block owns 16 rows x 256 cols; wave -> 2 tiles
// =====================================================================
__global__ __launch_bounds__(256) void proj_kernel(
    const float* __restrict__ in, const __bf16* __restrict__ W,
    const float* __restrict__ bias, __bf16* __restrict__ outbf, float scale)
{
    const int rowbase = blockIdx.x * 16;
    const int wave = threadIdx.x >> 5;
    const int lane = threadIdx.x & 31;
    const int lrow = lane & 15;

#pragma unroll
    for (int t = 0; t < 2; ++t) {
        const int cbase = (wave * 2 + t) * 16;
        v8f acc = {};
        for (int kb = 0; kb < HH; kb += 32) {
            v16bf a, b;
            const float*  arow = in + (size_t)(rowbase + lrow) * HH + kb;
            const __bf16* wrow = W  + (size_t)(cbase  + lrow) * HH + kb;
#pragma unroll
            for (int j = 0; j < 16; ++j) {
                a[j] = (__bf16)arow[a_k(lane, j)];
                b[j] = wrow[b_k(lane, j)];
            }
            acc = wmma_bf16(a, b, acc);
        }
#pragma unroll
        for (int i = 0; i < 8; ++i) {
            const int r = c_row(lane, i);
            const int c = cbase + lrow;
            outbf[(size_t)(rowbase + r) * HH + c] = (__bf16)((acc[i] + bias[c]) * scale);
        }
    }
}

// =====================================================================
// Kernel 2: flash-style cross attention with async-LDS K/V staging.
// block = (batch, 16-query tile); 8 waves = 8 heads sharing the staged chunk.
// =====================================================================
__global__ __launch_bounds__(256) void attn_kernel(
    const __bf16* __restrict__ qbf, const __bf16* __restrict__ kbf,
    const __bf16* __restrict__ vbf, __bf16* __restrict__ ctxbf,
    float* __restrict__ attn_score)
{
    __shared__ __bf16 kchunk[32][HH];         // 16 KB: 32 scene rows, all heads
    __shared__ __bf16 vchunk[32][HH];         // 16 KB
    __shared__ float  pstage[NHEAD][16][32];  // 16 KB: per-head P chunk / head-mean src

    const int blk   = blockIdx.x;
    const int b     = blk / (MM / 16);
    const int mbase = (blk % (MM / 16)) * 16;
    const int head  = threadIdx.x >> 5;       // wave id
    const int lane  = threadIdx.x & 31;
    const int lrow  = lane & 15;
    const int lhalf = lane >> 4;
    const int dofs  = head * DH + (lhalf << 4);   // B-frag k slice within a K/V row

    // ---- Q tile as A fragment (16 x 32 bf16) ----
    const __bf16* qrow = qbf + (size_t)(b * MM + mbase + lrow) * HH + head * DH;
    v16bf qa;
#pragma unroll
    for (int j = 0; j < 16; ++j) qa[j] = qrow[a_k(lane, j)];

    const __bf16* krows = kbf + (size_t)b * NS * HH;
    const __bf16* vrows = vbf + (size_t)b * NS * HH;

    // ---- pass 1: online (max, sumexp) per row; stats replicated in half-wave ----
    float rmax[8], rsum[8];
#pragma unroll
    for (int i = 0; i < 8; ++i) { rmax[i] = -1e30f; rsum[i] = 0.f; }

    for (int nb = 0; nb < NS; nb += 32) {
        load_chunk_async(kchunk, krows + (size_t)nb * HH, threadIdx.x);
        wait_async0();
        __syncthreads();                       // chunk staged for all waves

        v16bf kf0, kf1;
#pragma unroll
        for (int j = 0; j < 16; ++j) {
            kf0[j] = kchunk[lrow][dofs + j];        // cols nb..nb+15
            kf1[j] = kchunk[16 + lrow][dofs + j];   // cols nb+16..nb+31
        }
        v8f s0 = {}, s1 = {};
        s0 = wmma_bf16(qa, kf0, s0);
        s1 = wmma_bf16(qa, kf1, s1);
#pragma unroll
        for (int i = 0; i < 8; ++i) {
            const float cm = redmax16(fmaxf(s0[i], s1[i]));
            const float nm = fmaxf(rmax[i], cm);
            const float cs = redsum16(__expf(s0[i] - nm) + __expf(s1[i] - nm));
            rsum[i] = rsum[i] * __expf(rmax[i] - nm) + cs;
            rmax[i] = nm;
        }
        __syncthreads();                       // all waves done before restaging
    }
    float rinv[8];
#pragma unroll
    for (int i = 0; i < 8; ++i) rinv[i] = 1.0f / rsum[i];

    // ---- pass 2: P = softmax, ctx += P @ V, head-mean -> attn_score ----
    v8f c0 = {}, c1 = {};
    for (int nb = 0; nb < NS; nb += 32) {
        load_chunk_async(kchunk, krows + (size_t)nb * HH, threadIdx.x);
        load_chunk_async(vchunk, vrows + (size_t)nb * HH, threadIdx.x);
        wait_async0();
        __syncthreads();                       // K+V chunk staged

        v16bf kf0, kf1;
#pragma unroll
        for (int j = 0; j < 16; ++j) {
            kf0[j] = kchunk[lrow][dofs + j];
            kf1[j] = kchunk[16 + lrow][dofs + j];
        }
        v8f s0 = {}, s1 = {};
        s0 = wmma_bf16(qa, kf0, s0);
        s1 = wmma_bf16(qa, kf1, s1);

#pragma unroll
        for (int i = 0; i < 8; ++i) {
            const int r = c_row(lane, i);
            pstage[head][r][lrow]      = __expf(s0[i] - rmax[i]) * rinv[i];
            pstage[head][r][16 + lrow] = __expf(s1[i] - rmax[i]) * rinv[i];
        }
        __syncthreads();                       // all heads' P chunks visible

        // deterministic head-mean write of this 32-column chunk (512 elems / 256 thr)
        for (int idx = threadIdx.x; idx < 16 * 32; idx += 256) {
            const int r = idx >> 5, c = idx & 31;
            float sacc = 0.f;
#pragma unroll
            for (int h = 0; h < NHEAD; ++h) sacc += pstage[h][r][c];
            attn_score[(size_t)(b * MM + mbase + r) * NS + nb + c] = sacc * 0.125f;
        }

        // re-layout P: C-frag -> A-frag (16x32) through LDS
        v16bf pa;
#pragma unroll
        for (int j = 0; j < 16; ++j) pa[j] = (__bf16)pstage[head][lrow][a_k(lane, j)];

        // V chunk as two B fragments (32 x 16 each): d = 0..15 and 16..31
        v16bf vb0, vb1;
#pragma unroll
        for (int j = 0; j < 16; ++j) {
            const int n = b_k(lane, j);
            vb0[j] = vchunk[n][head * DH + lrow];
            vb1[j] = vchunk[n][head * DH + 16 + lrow];
        }
        c0 = wmma_bf16(pa, vb0, c0);
        c1 = wmma_bf16(pa, vb1, c1);
        __syncthreads();                       // all reads done before restaging
    }

    // ---- write ctx tile (16 x 32) as bf16 ----
#pragma unroll
    for (int i = 0; i < 8; ++i) {
        const int r = c_row(lane, i);
        const size_t off = (size_t)(b * MM + mbase + r) * HH + head * DH;
        ctxbf[off + lrow]      = (__bf16)c0[i];
        ctxbf[off + 16 + lrow] = (__bf16)c1[i];
    }
}

// =====================================================================
// Kernel 3: x = LayerNorm(ctx @ Wo^T + bo + hidden)   (Wo already bf16)
// =====================================================================
__global__ __launch_bounds__(256) void outproj_ln_kernel(
    const __bf16* __restrict__ ctxbf, const __bf16* __restrict__ Wo,
    const float* __restrict__ bo, const float* __restrict__ hidden,
    const float* __restrict__ gam, const float* __restrict__ bet,
    float* __restrict__ xout)
{
    __shared__ float xbuf[16][HH];
    __shared__ float mu[16], rstd[16];

    const int rowbase = blockIdx.x * 16;
    const int wave = threadIdx.x >> 5;
    const int lane = threadIdx.x & 31;
    const int lrow = lane & 15;

#pragma unroll
    for (int t = 0; t < 2; ++t) {
        const int cbase = (wave * 2 + t) * 16;
        v8f acc = {};
        for (int kb = 0; kb < HH; kb += 32) {
            v16bf a, b;
            const __bf16* arow = ctxbf + (size_t)(rowbase + lrow) * HH + kb;
            const __bf16* wrow = Wo    + (size_t)(cbase  + lrow) * HH + kb;
#pragma unroll
            for (int j = 0; j < 16; ++j) {
                a[j] = arow[a_k(lane, j)];
                b[j] = wrow[b_k(lane, j)];
            }
            acc = wmma_bf16(a, b, acc);
        }
#pragma unroll
        for (int i = 0; i < 8; ++i) {
            const int r = c_row(lane, i);
            const int c = cbase + lrow;
            xbuf[r][c] = acc[i] + bo[c] + hidden[(size_t)(rowbase + r) * HH + c];
        }
    }
    __syncthreads();

    if (threadIdx.x < 16) {
        const int r = threadIdx.x;
        float s = 0.f;
        for (int c = 0; c < HH; ++c) s += xbuf[r][c];
        const float mean = s * (1.0f / HH);
        float var = 0.f;
        for (int c = 0; c < HH; ++c) { float d = xbuf[r][c] - mean; var += d * d; }
        mu[r] = mean;
        rstd[r] = rsqrtf(var * (1.0f / HH) + LN_EPS);
    }
    __syncthreads();

    {
        const int r  = threadIdx.x >> 4;
        const int c0 = (threadIdx.x & 15) * 16;
#pragma unroll
        for (int i = 0; i < 16; ++i) {
            const int c = c0 + i;
            xout[(size_t)(rowbase + r) * HH + c] =
                (xbuf[r][c] - mu[r]) * rstd[r] * gam[c] + bet[c];
        }
    }
}

// =====================================================================
// Kernel 4: out = LayerNorm(relu(x @ W1^T + b1) @ W2^T + b2 + x)  (W1/W2 bf16)
// =====================================================================
__global__ __launch_bounds__(256) void ffn_ln_kernel(
    const float* __restrict__ x, const __bf16* __restrict__ W1,
    const float* __restrict__ b1, const __bf16* __restrict__ W2,
    const float* __restrict__ b2, const float* __restrict__ gam,
    const float* __restrict__ bet, float* __restrict__ out)
{
    __shared__ __bf16 h1buf[16][FFD];
    __shared__ float  ybuf[16][HH];
    __shared__ float  mu[16], rstd[16];

    const int rowbase = blockIdx.x * 16;
    const int wave = threadIdx.x >> 5;
    const int lane = threadIdx.x & 31;
    const int lrow = lane & 15;

    // ---- stage 1: h1 = relu(x @ W1^T + b1), one 16x16 tile per wave ----
    {
        const int cbase = wave * 16;
        v8f acc = {};
        for (int kb = 0; kb < HH; kb += 32) {
            v16bf a, b;
            const float*  arow = x  + (size_t)(rowbase + lrow) * HH + kb;
            const __bf16* wrow = W1 + (size_t)(cbase  + lrow) * HH + kb;
#pragma unroll
            for (int j = 0; j < 16; ++j) {
                a[j] = (__bf16)arow[a_k(lane, j)];
                b[j] = wrow[b_k(lane, j)];
            }
            acc = wmma_bf16(a, b, acc);
        }
#pragma unroll
        for (int i = 0; i < 8; ++i) {
            const int r = c_row(lane, i);
            const int c = cbase + lrow;
            h1buf[r][c] = (__bf16)fmaxf(acc[i] + b1[c], 0.0f);
        }
    }
    __syncthreads();

    // ---- stage 2: y = h1 @ W2^T + b2 + x, two tiles per wave, K=128 ----
#pragma unroll
    for (int t = 0; t < 2; ++t) {
        const int cbase = (wave * 2 + t) * 16;
        v8f acc = {};
        for (int kb = 0; kb < FFD; kb += 32) {
            v16bf a, b;
            const __bf16* wrow = W2 + (size_t)(cbase + lrow) * FFD + kb;
#pragma unroll
            for (int j = 0; j < 16; ++j) {
                a[j] = h1buf[lrow][kb + a_k(lane, j)];
                b[j] = wrow[b_k(lane, j)];
            }
            acc = wmma_bf16(a, b, acc);
        }
#pragma unroll
        for (int i = 0; i < 8; ++i) {
            const int r = c_row(lane, i);
            const int c = cbase + lrow;
            ybuf[r][c] = acc[i] + b2[c] + x[(size_t)(rowbase + r) * HH + c];
        }
    }
    __syncthreads();

    if (threadIdx.x < 16) {
        const int r = threadIdx.x;
        float s = 0.f;
        for (int c = 0; c < HH; ++c) s += ybuf[r][c];
        const float mean = s * (1.0f / HH);
        float var = 0.f;
        for (int c = 0; c < HH; ++c) { float d = ybuf[r][c] - mean; var += d * d; }
        mu[r] = mean;
        rstd[r] = rsqrtf(var * (1.0f / HH) + LN_EPS);
    }
    __syncthreads();

    {
        const int r  = threadIdx.x >> 4;
        const int c0 = (threadIdx.x & 15) * 16;
#pragma unroll
        for (int i = 0; i < 16; ++i) {
            const int c = c0 + i;
            out[(size_t)(rowbase + r) * HH + c] =
                (ybuf[r][c] - mu[r]) * rstd[r] * gam[c] + bet[c];
        }
    }
}

// =====================================================================
extern "C" void kernel_launch(void* const* d_in, const int* in_sizes, int n_in,
                              void* d_out, int out_size, void* d_ws, size_t ws_size,
                              hipStream_t stream)
{
    (void)in_sizes; (void)n_in; (void)out_size; (void)ws_size;

    const float* hidden = (const float*)d_in[0];   // (B, M, H)
    const float* scene  = (const float*)d_in[1];   // (B, NS, H)
    const float* Wqkv   = (const float*)d_in[2];   // (3H, H)
    const float* bqkv   = (const float*)d_in[3];   // (3H)
    const float* Wo     = (const float*)d_in[4];   // (H, H)
    const float* bo     = (const float*)d_in[5];
    const float* ag     = (const float*)d_in[6];
    const float* ab     = (const float*)d_in[7];
    const float* W1     = (const float*)d_in[8];   // (FF, H)
    const float* b1     = (const float*)d_in[9];
    const float* W2     = (const float*)d_in[10];  // (H, FF)
    const float* b2     = (const float*)d_in[11];
    const float* fg     = (const float*)d_in[12];
    const float* fb     = (const float*)d_in[13];

    float* out_main   = (float*)d_out;                          // (B,A,T,NMODE,H)
    float* attn_score = (float*)d_out + (size_t)BB * MM * HH;   // (B,M,1,NS)

    // ---- workspace layout ----
    char* ws = (char*)d_ws;
    size_t off = 0;
    __bf16* qbf    = (__bf16*)(ws + off); off += (size_t)BB * MM * HH * 2;
    __bf16* kbf    = (__bf16*)(ws + off); off += (size_t)BB * NS * HH * 2;
    __bf16* vbf    = (__bf16*)(ws + off); off += (size_t)BB * NS * HH * 2;
    __bf16* ctxbf  = (__bf16*)(ws + off); off += (size_t)BB * MM * HH * 2;
    float*  xws    = (float*) (ws + off); off += (size_t)BB * MM * HH * 4;
    __bf16* wqkvbf = (__bf16*)(ws + off); off += (size_t)3 * HH * HH * 2;
    __bf16* wobf   = (__bf16*)(ws + off); off += (size_t)HH * HH * 2;
    __bf16* w1bf   = (__bf16*)(ws + off); off += (size_t)FFD * HH * 2;
    __bf16* w2bf   = (__bf16*)(ws + off); off += (size_t)HH * FFD * 2;

    const float qscale = 0.17677669529663687f;  // 1/sqrt(DH)

    // one-shot weight conversion to bf16
    cvt_bf16_kernel<<<(3 * HH * HH + 255) / 256, 256, 0, stream>>>(Wqkv, wqkvbf, 3 * HH * HH);
    cvt_bf16_kernel<<<(HH * HH + 255) / 256, 256, 0, stream>>>(Wo, wobf, HH * HH);
    cvt_bf16_kernel<<<(FFD * HH + 255) / 256, 256, 0, stream>>>(W1, w1bf, FFD * HH);
    cvt_bf16_kernel<<<(HH * FFD + 255) / 256, 256, 0, stream>>>(W2, w2bf, HH * FFD);

    proj_kernel<<<(BB * MM) / 16, 256, 0, stream>>>(hidden, wqkvbf, bqkv, qbf, qscale);
    proj_kernel<<<(BB * NS) / 16, 256, 0, stream>>>(scene, wqkvbf + (size_t)HH * HH,
                                                    bqkv + HH, kbf, 1.0f);
    proj_kernel<<<(BB * NS) / 16, 256, 0, stream>>>(scene, wqkvbf + (size_t)2 * HH * HH,
                                                    bqkv + 2 * HH, vbf, 1.0f);
    attn_kernel<<<(BB * MM) / 16, 256, 0, stream>>>(qbf, kbf, vbf, ctxbf, attn_score);
    outproj_ln_kernel<<<(BB * MM) / 16, 256, 0, stream>>>(ctxbf, wobf, bo, hidden, ag, ab, xws);
    ffn_ln_kernel<<<(BB * MM) / 16, 256, 0, stream>>>(xws, w1bf, b1, w2bf, b2, fg, fb, out_main);
}
